// Causal_41120016892149
// MI455X (gfx1250) — compile-verified, matrix operand order checked by reading, not run on hardware
//
#include <hip/hip_runtime.h>
#include <stdint.h>

#define NN 50000
#define EE 800000
#define DD 128
#define LL 3
#define BB 512
#define EPS 1e-5f

typedef __attribute__((ext_vector_type(16))) __bf16   v16bf;
typedef __attribute__((ext_vector_type(8)))  float    v8f;
typedef __attribute__((ext_vector_type(4)))  uint32_t u32x4;
typedef __attribute__((ext_vector_type(4)))  float    f32x4;

union Frag16 { u32x4 q[2]; v16bf v; };

__device__ __forceinline__ uint16_t f2bf(float f) {
  uint32_t u = __float_as_uint(f);
  uint32_t r = u + 0x7FFFu + ((u >> 16) & 1u);   // round-to-nearest-even
  return (uint16_t)(r >> 16);
}

__device__ __forceinline__ float wave_sum(float v) {
  #pragma unroll
  for (int m = 16; m >= 1; m >>= 1) v += __shfl_xor(v, m, 32);
  return v;
}

// ---------------------------------------------------------------- utility
__global__ void zero_f32_kernel(float* p, long long n) {
  long long i = (long long)blockIdx.x * blockDim.x + threadIdx.x;
  if (i < n) p[i] = 0.f;
}

// agg[col[e]] += h[row[e]]  (thread per (edge, 4-float chunk))
__global__ void scatter_add_edges_kernel(const float* __restrict__ h,
                                         const int* __restrict__ row,
                                         const int* __restrict__ col,
                                         float* __restrict__ agg) {
  long long t = (long long)blockIdx.x * blockDim.x + threadIdx.x;
  if (t >= (long long)EE * 32) return;
  int e  = (int)(t >> 5);
  int d4 = (int)(t & 31);
  int r = row[e], c = col[e];
  f32x4 v = reinterpret_cast<const f32x4*>(h + (size_t)r * DD)[d4];
  float* dst = agg + (size_t)c * DD + d4 * 4;
  atomicAdd(dst + 0, v.x); atomicAdd(dst + 1, v.y);
  atomicAdd(dst + 2, v.z); atomicAdd(dst + 3, v.w);
}

// zb = bf16(h + agg)
__global__ void pack_zb_kernel(const float* __restrict__ h, const float* __restrict__ agg,
                               uint16_t* __restrict__ zb, long long n) {
  long long i = (long long)blockIdx.x * blockDim.x + threadIdx.x;
  if (i < n) zb[i] = f2bf(h[i] + agg[i]);
}

// fold linear-bias + BatchNorm(eval) into per-column scale/shift
__global__ void bn_prep_kernel(const float* __restrict__ bias, const float* __restrict__ g,
                               const float* __restrict__ b, const float* __restrict__ m,
                               const float* __restrict__ v, float* __restrict__ scale,
                               float* __restrict__ shift, int n) {
  int i = blockIdx.x * blockDim.x + threadIdx.x;
  if (i < n) {
    float s = g[i] * rsqrtf(v[i] + EPS);
    scale[i] = s;
    shift[i] = (bias[i] - m[i]) * s + b[i];
  }
}

// Pack fp32 KxN weight into per-wave bf16 B fragments mirroring the 16-bit A layout:
// out[((kt*NT+nt)*32+lane)*8+j] = pair {B[k][n], B[k+1][n]},
//   n = nt*16+(lane&15),  k = kt*32 + (j<4 ? 2j : 16+2(j-4)) + (lane>=16 ? 8 : 0)
__global__ void pack_weight_kernel(const float* __restrict__ W, int K, int Nn,
                                   uint32_t* __restrict__ out) {
  int idx = blockIdx.x * blockDim.x + threadIdx.x;
  int NT = Nn >> 4;
  int total = (K >> 5) * NT * 256;
  if (idx >= total) return;
  int j    = idx & 7;
  int lane = (idx >> 3) & 31;
  int ktnt = idx >> 8;
  int nt = ktnt % NT, kt = ktnt / NT;
  int n  = nt * 16 + (lane & 15);
  int kl = (j < 4 ? 2 * j : 16 + 2 * (j - 4)) + ((lane >> 4) << 3);
  int k  = kt * 32 + kl;
  uint32_t lo = f2bf(W[(size_t)k * Nn + n]);
  uint32_t hi = f2bf(W[(size_t)(k + 1) * Nn + n]);
  out[idx] = lo | (hi << 16);
}

// ---------------------------------------------------------------- WMMA GEMM
// One wave computes a 16x16 f32 tile of  A(M x K, bf16 row-major) * Bfrag(K x 16*NT).
// MODE 0: val=acc*scale+shift, relu, store bf16    (GIN layer-1 -> y1b)
// MODE 1: val=acc*scale+shift, RELU?, store f32    (GIN layer-2 -> h)
// MODE 2: raw f32                                  (GCN xw)
template <int KT, int MODE, int RELU>
__global__ __launch_bounds__(128)
void wmma_gemm_kernel(const uint16_t* __restrict__ A,
                      const uint32_t* __restrict__ Bfrag, int NT, int Nld,
                      const float* __restrict__ scale, const float* __restrict__ shift,
                      uint16_t* __restrict__ outb, float* __restrict__ outf) {
  const int K = KT * 32;
  int lane = threadIdx.x & 31;
  int wave = threadIdx.x >> 5;
  int nt = blockIdx.y * 4 + wave;
  if (nt >= NT) return;
  int m0 = blockIdx.x * 16;

  int mrow  = m0 + (lane & 15);
  int khalf = (lane >> 4) << 3;                   // +8 K-offset for upper half-wave
  const uint16_t* arow = A + (size_t)mrow * K + khalf;
  const uint32_t* bbase = Bfrag + ((size_t)nt * 32 + lane) * 8;

  v8f acc = {};
  #pragma unroll
  for (int kt = 0; kt < KT; ++kt) {
    Frag16 fa, fb;
    const u32x4* ap = reinterpret_cast<const u32x4*>(arow + kt * 32);
    fa.q[0] = ap[0];                              // K pairs {0..7}  rel. to khalf
    fa.q[1] = ap[2];                              // K pairs {16..23}
    const u32x4* bp = reinterpret_cast<const u32x4*>(bbase + (size_t)kt * NT * 256);
    fb.q[0] = bp[0];
    fb.q[1] = bp[1];
    acc = __builtin_amdgcn_wmma_f32_16x16x32_bf16(false, fa.v, false, fb.v,
                                                  (short)0, acc, false, false);
  }

  int col = nt * 16 + (lane & 15);
  float sc = 1.f, sh = 0.f;
  if (MODE != 2) { sc = scale[col]; sh = shift[col]; }
  int rbase = m0 + ((lane >> 4) << 3);            // C layout: vgpr v -> row v (+8 hi half)
  #pragma unroll
  for (int v = 0; v < 8; ++v) {
    float val = acc[v] * sc + sh;
    if (RELU) val = fmaxf(val, 0.f);
    size_t off = (size_t)(rbase + v) * Nld + col;
    if (MODE == 0) outb[off] = f2bf(val);
    else           outf[off] = val;
  }
}

// ---------------------------------------------------------------- attention
// per-node (wave): s_top = h.ew_W[:128], s_bot = h.ew_W[128:], nlog = h.nw_W
__global__ __launch_bounds__(256)
void node_attn_kernel(const float* __restrict__ h, const float* __restrict__ ewW,
                      const float* __restrict__ nwW, float* __restrict__ stp,
                      float* __restrict__ sbt, float* __restrict__ nlg) {
  int wid  = (int)(((long long)blockIdx.x * blockDim.x + threadIdx.x) >> 5);
  int lane = threadIdx.x & 31;
  if (wid >= NN) return;
  const float* hr = h + (size_t)wid * DD;
  float a0 = 0, a1 = 0, b0 = 0, b1 = 0, c0 = 0, c1 = 0;
  #pragma unroll
  for (int k = lane; k < DD; k += 32) {
    float x = hr[k];
    a0 += x * ewW[k * 2 + 0];          a1 += x * ewW[k * 2 + 1];
    b0 += x * ewW[(DD + k) * 2 + 0];   b1 += x * ewW[(DD + k) * 2 + 1];
    c0 += x * nwW[k * 2 + 0];          c1 += x * nwW[k * 2 + 1];
  }
  a0 = wave_sum(a0); a1 = wave_sum(a1);
  b0 = wave_sum(b0); b1 = wave_sum(b1);
  c0 = wave_sum(c0); c1 = wave_sum(c1);
  if (lane == 0) {
    stp[wid * 2] = a0; stp[wid * 2 + 1] = a1;
    sbt[wid * 2] = b0; sbt[wid * 2 + 1] = b1;
    nlg[wid * 2] = c0; nlg[wid * 2 + 1] = c1;
  }
}

// softmax over 2 logits, take class 1  ->  sigmoid(l1 - l0)
__global__ void edge_weight_kernel(const float* __restrict__ stp, const float* __restrict__ sbt,
                                   const float* __restrict__ ewb, const int* __restrict__ row,
                                   const int* __restrict__ col, float* __restrict__ ew) {
  int e = blockIdx.x * blockDim.x + threadIdx.x;
  if (e >= EE) return;
  int r = row[e], c = col[e];
  float l0 = stp[r * 2 + 0] + sbt[c * 2 + 0] + ewb[0];
  float l1 = stp[r * 2 + 1] + sbt[c * 2 + 1] + ewb[1];
  ew[e] = 1.f / (1.f + expf(l0 - l1));
}

__global__ void node_weight_kernel(const float* __restrict__ nlg, const float* __restrict__ nwb,
                                   float* __restrict__ nw) {
  int n = blockIdx.x * blockDim.x + threadIdx.x;
  if (n >= NN) return;
  float l0 = nlg[n * 2 + 0] + nwb[0];
  float l1 = nlg[n * 2 + 1] + nwb[1];
  nw[n] = 1.f / (1.f + expf(l0 - l1));
}

// causal = LayerNorm(nw[n] * h[n])  -> bf16 (wave per node, 4 elems/lane)
__global__ __launch_bounds__(256)
void causal_ln_kernel(const float* __restrict__ h, const float* __restrict__ nw,
                      const float* __restrict__ g, const float* __restrict__ b,
                      uint16_t* __restrict__ out) {
  int wid  = (int)(((long long)blockIdx.x * blockDim.x + threadIdx.x) >> 5);
  int lane = threadIdx.x & 31;
  if (wid >= NN) return;
  const f32x4* hr = reinterpret_cast<const f32x4*>(h + (size_t)wid * DD);
  float w = nw[wid];
  f32x4 x = hr[lane];
  float v0 = w * x.x, v1 = w * x.y, v2 = w * x.z, v3 = w * x.w;
  float s  = wave_sum(v0 + v1 + v2 + v3);
  float ss = wave_sum(v0 * v0 + v1 * v1 + v2 * v2 + v3 * v3);
  float mu = s * (1.f / DD);
  float rstd = rsqrtf(ss * (1.f / DD) - mu * mu + EPS);
  int d = lane * 4;
  uint16_t* o = out + (size_t)wid * DD + d;
  o[0] = f2bf((v0 - mu) * rstd * g[d + 0] + b[d + 0]);
  o[1] = f2bf((v1 - mu) * rstd * g[d + 1] + b[d + 1]);
  o[2] = f2bf((v2 - mu) * rstd * g[d + 2] + b[d + 2]);
  o[3] = f2bf((v3 - mu) * rstd * g[d + 3] + b[d + 3]);
}

// ---------------------------------------------------------------- GCN
__global__ void deg_scatter_kernel(const float* __restrict__ ew, const int* __restrict__ col,
                                   float* __restrict__ deg) {
  int e = blockIdx.x * blockDim.x + threadIdx.x;
  if (e < EE) atomicAdd(deg + col[e], ew[e]);
}

__global__ void dinv_kernel(float* __restrict__ deg) {
  int n = blockIdx.x * blockDim.x + threadIdx.x;
  if (n < NN) deg[n] = rsqrtf(deg[n] + 1.f);     // +1 from self-loop; always > 0
}

__global__ void gcn_scatter_kernel(const float* __restrict__ xw, const float* __restrict__ dinv,
                                   const float* __restrict__ ew, const int* __restrict__ row,
                                   const int* __restrict__ col, float* __restrict__ conv) {
  long long t = (long long)blockIdx.x * blockDim.x + threadIdx.x;
  if (t >= (long long)EE * 32) return;
  int e  = (int)(t >> 5);
  int d4 = (int)(t & 31);
  int r = row[e], c = col[e];
  float coef = dinv[r] * ew[e] * dinv[c];
  f32x4 v = reinterpret_cast<const f32x4*>(xw + (size_t)r * DD)[d4];
  float* dst = conv + (size_t)c * DD + d4 * 4;
  atomicAdd(dst + 0, coef * v.x); atomicAdd(dst + 1, coef * v.y);
  atomicAdd(dst + 2, coef * v.z); atomicAdd(dst + 3, coef * v.w);
}

// conv = relu(conv + dinv^2 * xw + bias)   (self-loop term + bias + relu)
__global__ void gcn_finalize_kernel(float* __restrict__ conv, const float* __restrict__ xw,
                                    const float* __restrict__ dinv, const float* __restrict__ bias) {
  long long i = (long long)blockIdx.x * blockDim.x + threadIdx.x;
  if (i >= (long long)NN * DD) return;
  int d = (int)(i & (DD - 1));
  int n = (int)(i >> 7);
  float di = dinv[n];
  conv[i] = fmaxf(conv[i] + di * di * xw[i] + bias[d], 0.f);
}

__global__ void pool_scatter_kernel(const float* __restrict__ conv, const int* __restrict__ batch,
                                    float* __restrict__ pooled) {
  long long t = (long long)blockIdx.x * blockDim.x + threadIdx.x;
  if (t >= (long long)NN * 32) return;
  int n  = (int)(t >> 5);
  int d4 = (int)(t & 31);
  int bi = batch[n];
  f32x4 v = reinterpret_cast<const f32x4*>(conv + (size_t)n * DD)[d4];
  float* dst = pooled + (size_t)bi * DD + d4 * 4;
  atomicAdd(dst + 0, v.x); atomicAdd(dst + 1, v.y);
  atomicAdd(dst + 2, v.z); atomicAdd(dst + 3, v.w);
}

// ---------------------------------------------------------------- readout
__global__ __launch_bounds__(256)
void ln_rows_kernel(const float* __restrict__ in, const float* __restrict__ g,
                    const float* __restrict__ b, float* __restrict__ out, int rows) {
  int wid  = (int)(((long long)blockIdx.x * blockDim.x + threadIdx.x) >> 5);
  int lane = threadIdx.x & 31;
  if (wid >= rows) return;
  f32x4 x = reinterpret_cast<const f32x4*>(in + (size_t)wid * DD)[lane];
  float s  = wave_sum(x.x + x.y + x.z + x.w);
  float ss = wave_sum(x.x * x.x + x.y * x.y + x.z * x.z + x.w * x.w);
  float mu = s * (1.f / DD);
  float rstd = rsqrtf(ss * (1.f / DD) - mu * mu + EPS);
  int d = lane * 4;
  float* o = out + (size_t)wid * DD + d;
  o[0] = (x.x - mu) * rstd * g[d + 0] + b[d + 0];
  o[1] = (x.y - mu) * rstd * g[d + 1] + b[d + 1];
  o[2] = (x.z - mu) * rstd * g[d + 2] + b[d + 2];
  o[3] = (x.w - mu) * rstd * g[d + 3] + b[d + 3];
}

__global__ __launch_bounds__(128)
void mlp_sigmoid_kernel(const float* __restrict__ in, const float* __restrict__ W,
                        const float* __restrict__ bias, float* __restrict__ out) {
  __shared__ float s[DD];
  int r = blockIdx.x, t = threadIdx.x;
  s[t] = in[(size_t)r * DD + t];
  __syncthreads();
  float acc = bias[t];
  #pragma unroll 8
  for (int k = 0; k < DD; ++k) acc += s[k] * W[k * DD + t];
  out[(size_t)r * DD + t] = 1.f / (1.f + expf(-acc));
}

__global__ void final_proj_kernel(const float* __restrict__ in, const float* __restrict__ W,
                                  const float* __restrict__ bias, float* __restrict__ out) {
  int idx = blockIdx.x * blockDim.x + threadIdx.x;
  if (idx >= BB * 2) return;
  int r = idx >> 1, c = idx & 1;
  float acc = bias[c];
  #pragma unroll 8
  for (int k = 0; k < DD; ++k) acc += in[(size_t)r * DD + k] * W[k * 2 + c];
  out[idx] = acc;
}

// ================================================================ launch
static inline int gdiv(long long n, int b) { return (int)((n + b - 1) / b); }

extern "C" void kernel_launch(void* const* d_in, const int* in_sizes, int n_in,
                              void* d_out, int out_size, void* d_ws, size_t ws_size,
                              hipStream_t stream) {
  const float* x      = (const float*)d_in[0];
  const int*   row    = (const int*)d_in[1];
  const int*   col    = row + EE;
  const int*   batch  = (const int*)d_in[2];
  const float* gin_W1 = (const float*)d_in[3];
  const float* gin_b1 = (const float*)d_in[4];
  const float* bnm_g  = (const float*)d_in[5];
  const float* bnm_b  = (const float*)d_in[6];
  const float* bnm_m  = (const float*)d_in[7];
  const float* bnm_v  = (const float*)d_in[8];
  const float* gin_W2 = (const float*)d_in[9];
  const float* gin_b2 = (const float*)d_in[10];
  const float* bno_g  = (const float*)d_in[11];
  const float* bno_b  = (const float*)d_in[12];
  const float* bno_m  = (const float*)d_in[13];
  const float* bno_v  = (const float*)d_in[14];
  const float* ew_W   = (const float*)d_in[15];
  const float* ew_b   = (const float*)d_in[16];
  const float* nw_W   = (const float*)d_in[17];
  const float* nw_b   = (const float*)d_in[18];
  const float* ln_g   = (const float*)d_in[19];
  const float* ln_b   = (const float*)d_in[20];
  const float* gcn_W  = (const float*)d_in[21];
  const float* gcn_b  = (const float*)d_in[22];
  const float* m_ln1g = (const float*)d_in[23];
  const float* m_ln1b = (const float*)d_in[24];
  const float* m_W1   = (const float*)d_in[25];
  const float* m_b1   = (const float*)d_in[26];
  const float* m_ln2g = (const float*)d_in[27];
  const float* m_ln2b = (const float*)d_in[28];
  const float* m_W2   = (const float*)d_in[29];
  const float* m_b2   = (const float*)d_in[30];
  float* out = (float*)d_out;

  // ---- workspace partition (bytes) ----
  char* p = (char*)d_ws;
  float*    h      = (float*)p;              p += (size_t)NN * DD * 4;   // 25.6 MB
  float*    agg    = (float*)p;              p += (size_t)NN * DD * 4;   // 25.6 MB (xw later)
  uint16_t* zb     = (uint16_t*)p;           p += (size_t)NN * DD * 2;   // 12.8 MB (causal later)
  uint16_t* y1b    = (uint16_t*)p;           p += (size_t)NN * 2 * DD * 2; // 25.6 MB (conv later)
  uint32_t* w1f    = (uint32_t*)p;           p += 16384 * 4;
  uint32_t* w2f    = (uint32_t*)p;           p += 16384 * 4;
  uint32_t* wgf    = (uint32_t*)p;           p += 8192 * 4;
  float*    scale1 = (float*)p;              p += 256 * 4;
  float*    shift1 = (float*)p;              p += 256 * 4;
  float*    scale2 = (float*)p;              p += 128 * 4;
  float*    shift2 = (float*)p;              p += 128 * 4;
  float*    stp    = (float*)p;              p += (size_t)NN * 2 * 4;
  float*    sbt    = (float*)p;              p += (size_t)NN * 2 * 4;
  float*    nlg    = (float*)p;              p += (size_t)NN * 2 * 4;
  float*    ew     = (float*)p;              p += (size_t)EE * 4;
  float*    nwv    = (float*)p;              p += (size_t)NN * 4;
  float*    deg    = (float*)p;              p += (size_t)NN * 4;        // becomes dinv
  float*    pooled = (float*)p;              p += (size_t)BB * DD * 4;
  float*    o1     = (float*)p;              p += (size_t)BB * DD * 4;
  float*    o2     = (float*)p;              p += (size_t)BB * DD * 4;
  float*    xw   = agg;            // reuse after GIN layers
  float*    conv = (float*)y1b;    // reuse after GIN layers (same byte size)

  const long long nd  = (long long)NN * DD;
  const long long e32 = (long long)EE * 32;
  const int MT = NN / 16;          // 3125 row tiles (N divisible by 16)

  // ---- GIN encoder: 3 layers ----
  const float* hcur = x;
  for (int i = 0; i < LL; ++i) {
    zero_f32_kernel<<<gdiv(nd, 256), 256, 0, stream>>>(agg, nd);
    scatter_add_edges_kernel<<<gdiv(e32, 256), 256, 0, stream>>>(hcur, row, col, agg);
    pack_zb_kernel<<<gdiv(nd, 256), 256, 0, stream>>>(hcur, agg, zb, nd);

    bn_prep_kernel<<<1, 256, 0, stream>>>(gin_b1 + i * 256, bnm_g + i * 256, bnm_b + i * 256,
                                          bnm_m + i * 256, bnm_v + i * 256, scale1, shift1, 256);
    pack_weight_kernel<<<gdiv(4 * 16 * 256, 256), 256, 0, stream>>>(
        gin_W1 + (size_t)i * 128 * 256, 128, 256, w1f);
    // y1b = relu(bn(zb @ W1))  [N x 256] bf16
    wmma_gemm_kernel<4, 0, 1><<<dim3(MT, 4), 128, 0, stream>>>(
        zb, w1f, 16, 256, scale1, shift1, y1b, nullptr);

    bn_prep_kernel<<<1, 128, 0, stream>>>(gin_b2 + i * 128, bno_g + i * 128, bno_b + i * 128,
                                          bno_m + i * 128, bno_v + i * 128, scale2, shift2, 128);
    pack_weight_kernel<<<gdiv(8 * 8 * 256, 256), 256, 0, stream>>>(
        gin_W2 + (size_t)i * 256 * 128, 256, 128, w2f);
    // h = bn(y1b @ W2) (+relu except last layer)  [N x 128] f32
    if (i != LL - 1)
      wmma_gemm_kernel<8, 1, 1><<<dim3(MT, 2), 128, 0, stream>>>(
          y1b, w2f, 8, 128, scale2, shift2, nullptr, h);
    else
      wmma_gemm_kernel<8, 1, 0><<<dim3(MT, 2), 128, 0, stream>>>(
          y1b, w2f, 8, 128, scale2, shift2, nullptr, h);
    hcur = h;
  }

  // ---- edge / node attention ----
  node_attn_kernel<<<gdiv((long long)NN * 32, 256), 256, 0, stream>>>(h, ew_W, nw_W, stp, sbt, nlg);
  edge_weight_kernel<<<gdiv(EE, 256), 256, 0, stream>>>(stp, sbt, ew_b, row, col, ew);
  node_weight_kernel<<<gdiv(NN, 256), 256, 0, stream>>>(nlg, nw_b, nwv);
  causal_ln_kernel<<<gdiv((long long)NN * 32, 256), 256, 0, stream>>>(h, nwv, ln_g, ln_b, zb);

  // ---- GCNConv ----
  pack_weight_kernel<<<gdiv(4 * 8 * 256, 256), 256, 0, stream>>>(gcn_W, 128, 128, wgf);
  wmma_gemm_kernel<4, 2, 0><<<dim3(MT, 2), 128, 0, stream>>>(
      zb, wgf, 8, 128, nullptr, nullptr, nullptr, xw);
  zero_f32_kernel<<<gdiv(NN, 256), 256, 0, stream>>>(deg, NN);
  deg_scatter_kernel<<<gdiv(EE, 256), 256, 0, stream>>>(ew, col, deg);
  dinv_kernel<<<gdiv(NN, 256), 256, 0, stream>>>(deg);
  zero_f32_kernel<<<gdiv(nd, 256), 256, 0, stream>>>(conv, nd);
  gcn_scatter_kernel<<<gdiv(e32, 256), 256, 0, stream>>>(xw, deg, ew, row, col, conv);
  gcn_finalize_kernel<<<gdiv(nd, 256), 256, 0, stream>>>(conv, xw, deg, gcn_b);

  // ---- pool + readout MLP ----
  zero_f32_kernel<<<gdiv((long long)BB * DD, 256), 256, 0, stream>>>(pooled, (long long)BB * DD);
  pool_scatter_kernel<<<gdiv((long long)NN * 32, 256), 256, 0, stream>>>(conv, batch, pooled);
  ln_rows_kernel<<<gdiv((long long)BB * 32, 256), 256, 0, stream>>>(pooled, m_ln1g, m_ln1b, o1, BB);
  mlp_sigmoid_kernel<<<BB, 128, 0, stream>>>(o1, m_W1, m_b1, o2);
  ln_rows_kernel<<<gdiv((long long)BB * 32, 256), 256, 0, stream>>>(o2, m_ln2g, m_ln2b, o1, BB);
  final_proj_kernel<<<gdiv(BB * 2, 256), 256, 0, stream>>>(o1, m_W2, m_b2, out);
}
